// Attention_11639361372161
// MI455X (gfx1250) — matrix operand order repaired and checked by reading, not compiled, not run on hardware
//
#include <hip/hip_runtime.h>
#include <hip/hip_bf16.h>

#define B_    8
#define S_    785
#define D_    768
#define H_    12
#define HD_   64
#define TOPK_ 685

typedef __attribute__((ext_vector_type(16))) __bf16        v16bf;
typedef __attribute__((ext_vector_type(8)))  float         v8f;
typedef __attribute__((ext_vector_type(8)))  unsigned int  v8u;

static __device__ __forceinline__ unsigned short f2bf(float f) {
    unsigned u = __float_as_uint(f);
    u += 0x7FFFu + ((u >> 16) & 1u);   // round-to-nearest-even
    return (unsigned short)(u >> 16);
}
// split one f32 into hi/lo bf16 (x ~= hi + lo, 16 explicit mantissa bits)
static __device__ __forceinline__ void splitbf(float a, unsigned short& h, unsigned short& l) {
    h = f2bf(a);
    float fh = __uint_as_float((unsigned)h << 16);
    l = f2bf(a - fh);
}
static __device__ __forceinline__ void split2bf(float a, float b, unsigned& hi, unsigned& lo) {
    unsigned short ha, la, hb, lb;
    splitbf(a, ha, la);
    splitbf(b, hb, lb);
    hi = (unsigned)ha | ((unsigned)hb << 16);
    lo = (unsigned)la | ((unsigned)lb << 16);
}

static __device__ __forceinline__ v16bf frag_from(const uint4* base, int i0, int i1) {
    uint4 lo = base[i0];
    uint4 hi = base[i1];
    v8u r = { lo.x, lo.y, lo.z, lo.w, hi.x, hi.y, hi.z, hi.w };
    return __builtin_bit_cast(v16bf, r);
}

#define WMMA_BF16(A, Bv, Cv) \
    __builtin_amdgcn_wmma_f32_16x16x32_bf16(false, (A), false, (Bv), (short)0, (Cv), false, false)

// ---------------------------------------------------------------------------
// Generic batched emulated-fp32 WMMA GEMM via bf16 hi/lo 3-term split:
//   C[z] = scale * A[z] (.) B[z] + bias,   A.B ~= Ah.Bh + Ah.Bl + Al.Bh
//   A: M x K row-major (lda), per-batch offset zb*sAb + zh*sAh
//   B: if !transB, K x N row-major (ldb); if transB, N x K row-major (ldb)
// Block: 256 threads = 8 waves; tile 128x64, BK=32; wave -> 32x32 slab,
// 12 WMMAs per K-step.
// ---------------------------------------------------------------------------
__global__ __launch_bounds__(256)
void wmma_gemm(const float* __restrict__ A, const float* __restrict__ Bm,
               const float* __restrict__ bias, float* __restrict__ C,
               int M, int N, int K, int lda, int ldb, int ldc,
               long long sAb, long long sAh, long long sBb, long long sBh,
               long long sCb, long long sCh, int Hdim, float scale, int transB)
{
    __shared__ alignas(16) unsigned short Ah[128 * 32];  // m-major: [m*32+k]
    __shared__ alignas(16) unsigned short Al[128 * 32];
    __shared__ alignas(16) unsigned short Bh[64 * 32];   // n-major: [n*32+k]
    __shared__ alignas(16) unsigned short Bl[64 * 32];

    const int tid = threadIdx.x;
    const int z   = blockIdx.z;
    const int zb  = z / Hdim;
    const int zh  = z % Hdim;
    const float* Ap = A  + (size_t)zb * sAb + (size_t)zh * sAh;
    const float* Bp = Bm + (size_t)zb * sBb + (size_t)zh * sBh;
    float*       Cp = C  + (size_t)zb * sCb + (size_t)zh * sCh;

    const int blockM = blockIdx.y * 128;
    const int blockN = blockIdx.x * 64;

    const int wid  = tid >> 5;
    const int lane = tid & 31;
    const int half = lane >> 4;
    const int l15  = lane & 15;
    const int mOff = (wid & 3) * 32;
    const int nOff = (wid >> 2) * 32;

    v8f acc00 = {0.f,0.f,0.f,0.f,0.f,0.f,0.f,0.f};
    v8f acc01 = {0.f,0.f,0.f,0.f,0.f,0.f,0.f,0.f};
    v8f acc10 = {0.f,0.f,0.f,0.f,0.f,0.f,0.f,0.f};
    v8f acc11 = {0.f,0.f,0.f,0.f,0.f,0.f,0.f,0.f};

    const uint4* A4h = (const uint4*)Ah;
    const uint4* A4l = (const uint4*)Al;
    const uint4* B4h = (const uint4*)Bh;
    const uint4* B4l = (const uint4*)Bl;
    const int mr0 = mOff + l15;
    const int mr1 = mr0 + 16;
    const int nc0 = nOff + l15;
    const int nc1 = nc0 + 16;

    const bool fullM = (blockM + 128 <= M);
    const bool fullN = (blockN + 64  <= N);
    const bool aVec  = ((lda & 3) == 0);
    const bool bVec  = ((ldb & 3) == 0);

    for (int k0 = 0; k0 < K; k0 += 32) {
        const bool fullK = (k0 + 32 <= K);
        __syncthreads();

        // ================= stage A tile (128x32) =================
        if (fullM && fullK && aVec) {
            #pragma unroll
            for (int it = 0; it < 4; ++it) {
                int idx4 = tid + it * 256;             // 1024 float4 slots
                int r = idx4 >> 3, c = (idx4 & 7) * 4;
                float4 f = *(const float4*)(Ap + (size_t)(blockM + r) * lda + k0 + c);
                uint2 ph, pl;
                split2bf(f.x, f.y, ph.x, pl.x);
                split2bf(f.z, f.w, ph.y, pl.y);
                *(uint2*)(&Ah[r * 32 + c]) = ph;
                *(uint2*)(&Al[r * 32 + c]) = pl;
            }
        } else if (fullM && fullK) {
            #pragma unroll
            for (int it = 0; it < 16; ++it) {
                int idx = tid + it * 256;
                int r = idx >> 5, c = idx & 31;
                unsigned short h, l;
                splitbf(Ap[(size_t)(blockM + r) * lda + k0 + c], h, l);
                Ah[idx] = h; Al[idx] = l;
            }
        } else {
            #pragma unroll
            for (int it = 0; it < 16; ++it) {
                int idx = tid + it * 256;
                int r = idx >> 5, c = idx & 31;
                int m = blockM + r, k = k0 + c;
                float v = (m < M && k < K) ? Ap[(size_t)m * lda + k] : 0.f;
                unsigned short h, l;
                splitbf(v, h, l);
                Ah[idx] = h; Al[idx] = l;
            }
        }

        // ================= stage B tile (n-major 64x32) =================
        if (transB) {
            if (fullN && fullK && bVec) {
                #pragma unroll
                for (int it = 0; it < 2; ++it) {
                    int idx4 = tid + it * 256;          // 512 float4 slots
                    int n_ = idx4 >> 3, c = (idx4 & 7) * 4;
                    float4 f = *(const float4*)(Bp + (size_t)(blockN + n_) * ldb + k0 + c);
                    uint2 ph, pl;
                    split2bf(f.x, f.y, ph.x, pl.x);
                    split2bf(f.z, f.w, ph.y, pl.y);
                    *(uint2*)(&Bh[n_ * 32 + c]) = ph;
                    *(uint2*)(&Bl[n_ * 32 + c]) = pl;
                }
            } else {
                #pragma unroll
                for (int it = 0; it < 8; ++it) {
                    int idx = tid + it * 256;
                    int n_ = idx >> 5, c = idx & 31;
                    int n = blockN + n_, k = k0 + c;
                    float v = (n < N && k < K) ? Bp[(size_t)n * ldb + k] : 0.f;
                    unsigned short h, l;
                    splitbf(v, h, l);
                    Bh[idx] = h; Bl[idx] = l;
                }
            }
        } else {
            if (fullN && fullK && bVec) {
                #pragma unroll
                for (int it = 0; it < 2; ++it) {
                    int idx4 = tid + it * 256;          // 512 float4 slots
                    int k_ = idx4 >> 4, n4 = (idx4 & 15) * 4;
                    float4 f = *(const float4*)(Bp + (size_t)(k0 + k_) * ldb + blockN + n4);
                    unsigned short h, l;
                    splitbf(f.x, h, l); Bh[(n4+0)*32 + k_] = h; Bl[(n4+0)*32 + k_] = l;
                    splitbf(f.y, h, l); Bh[(n4+1)*32 + k_] = h; Bl[(n4+1)*32 + k_] = l;
                    splitbf(f.z, h, l); Bh[(n4+2)*32 + k_] = h; Bl[(n4+2)*32 + k_] = l;
                    splitbf(f.w, h, l); Bh[(n4+3)*32 + k_] = h; Bl[(n4+3)*32 + k_] = l;
                }
            } else {
                #pragma unroll
                for (int it = 0; it < 8; ++it) {
                    int idx = tid + it * 256;
                    int r = idx >> 6, c = idx & 63;     // r = k, c = n
                    int k = k0 + r, n = blockN + c;
                    float v = (k < K && n < N) ? Bp[(size_t)k * ldb + n] : 0.f;
                    unsigned short h, l;
                    splitbf(v, h, l);
                    Bh[c * 32 + r] = h; Bl[c * 32 + r] = l;
                }
            }
        }
        __syncthreads();

        // ---- fragments (A and B: lanes 0-15 K0-7/16-23, 16-31 +8)
        v16bf a0h = frag_from(A4h, mr0 * 4 + half, mr0 * 4 + 2 + half);
        v16bf a0l = frag_from(A4l, mr0 * 4 + half, mr0 * 4 + 2 + half);
        v16bf a1h = frag_from(A4h, mr1 * 4 + half, mr1 * 4 + 2 + half);
        v16bf a1l = frag_from(A4l, mr1 * 4 + half, mr1 * 4 + 2 + half);
        v16bf b0h = frag_from(B4h, nc0 * 4 + half, nc0 * 4 + 2 + half);
        v16bf b0l = frag_from(B4l, nc0 * 4 + half, nc0 * 4 + 2 + half);
        v16bf b1h = frag_from(B4h, nc1 * 4 + half, nc1 * 4 + 2 + half);
        v16bf b1l = frag_from(B4l, nc1 * 4 + half, nc1 * 4 + 2 + half);

        // ---- 3-term emulated fp32: hi*hi + hi*lo + lo*hi
        acc00 = WMMA_BF16(a0h, b0h, acc00);
        acc00 = WMMA_BF16(a0h, b0l, acc00);
        acc00 = WMMA_BF16(a0l, b0h, acc00);
        acc01 = WMMA_BF16(a0h, b1h, acc01);
        acc01 = WMMA_BF16(a0h, b1l, acc01);
        acc01 = WMMA_BF16(a0l, b1h, acc01);
        acc10 = WMMA_BF16(a1h, b0h, acc10);
        acc10 = WMMA_BF16(a1h, b0l, acc10);
        acc10 = WMMA_BF16(a1l, b0h, acc10);
        acc11 = WMMA_BF16(a1h, b1h, acc11);
        acc11 = WMMA_BF16(a1h, b1l, acc11);
        acc11 = WMMA_BF16(a1l, b1h, acc11);
    }

    // ---- epilogue: D VGPR r -> row r + 8*half; col = lane&15 within subtile
    const int gn0 = blockN + nc0;
    const int gn1 = blockN + nc1;
    const float bv0e = (bias && gn0 < N) ? bias[gn0] : 0.f;
    const float bv1e = (bias && gn1 < N) ? bias[gn1] : 0.f;
    #pragma unroll
    for (int r = 0; r < 8; ++r) {
        int m0 = blockM + mOff + r + half * 8;
        int m1 = m0 + 16;
        if (m0 < M) {
            if (gn0 < N) Cp[(size_t)m0 * ldc + gn0] = acc00[r] * scale + bv0e;
            if (gn1 < N) Cp[(size_t)m0 * ldc + gn1] = acc01[r] * scale + bv1e;
        }
        if (m1 < M) {
            if (gn0 < N) Cp[(size_t)m1 * ldc + gn0] = acc10[r] * scale + bv0e;
            if (gn1 < N) Cp[(size_t)m1 * ldc + gn1] = acc11[r] * scale + bv1e;
        }
    }
}

// ---------------------------------------------------------------------------
// wave32 butterfly block reductions (256 threads = 8 waves)
// ---------------------------------------------------------------------------
static __device__ __forceinline__ int block_sum_i(int v, int* sh, int tid) {
    #pragma unroll
    for (int off = 16; off > 0; off >>= 1) v += __shfl_xor(v, off, 32);
    if ((tid & 31) == 0) sh[tid >> 5] = v;
    __syncthreads();
    if (tid < 32) {
        int x = ((tid & 31) < 8) ? sh[tid & 31] : 0;
        #pragma unroll
        for (int off = 4; off > 0; off >>= 1) x += __shfl_xor(x, off, 32);
        if (tid == 0) sh[0] = x;
    }
    __syncthreads();
    int r = sh[0];
    __syncthreads();
    return r;
}
static __device__ __forceinline__ float block_sum_f(float v, float* sh, int tid) {
    #pragma unroll
    for (int off = 16; off > 0; off >>= 1) v += __shfl_xor(v, off, 32);
    if ((tid & 31) == 0) sh[tid >> 5] = v;
    __syncthreads();
    if (tid < 32) {
        float x = ((tid & 31) < 8) ? sh[tid & 31] : 0.f;
        #pragma unroll
        for (int off = 4; off > 0; off >>= 1) x += __shfl_xor(x, off, 32);
        if (tid == 0) sh[0] = x;
    }
    __syncthreads();
    float r = sh[0];
    __syncthreads();
    return r;
}
static __device__ __forceinline__ float block_max_f(float v, float* sh, int tid) {
    #pragma unroll
    for (int off = 16; off > 0; off >>= 1) v = fmaxf(v, __shfl_xor(v, off, 32));
    if ((tid & 31) == 0) sh[tid >> 5] = v;
    __syncthreads();
    if (tid < 32) {
        float x = ((tid & 31) < 8) ? sh[tid & 31] : -3.4e38f;
        #pragma unroll
        for (int off = 4; off > 0; off >>= 1) x = fmaxf(x, __shfl_xor(x, off, 32));
        if (tid == 0) sh[0] = x;
    }
    __syncthreads();
    float r = sh[0];
    __syncthreads();
    return r;
}

// ---------------------------------------------------------------------------
// Per-row top-k threshold + mask + softmax, in place on the scores/probs
// tensor (B,H,S,S). One 256-thread block per (b,h,q) row.
// Exact k-th largest via 32-step bit-greedy select on order-preserving uints.
// ---------------------------------------------------------------------------
__global__ __launch_bounds__(256)
void topk_softmax(float* __restrict__ scores, const float* __restrict__ maskfb,
                  const int* __restrict__ fcp, const int* __restrict__ lnp)
{
    __shared__ float    srow[S_];
    __shared__ unsigned urow[S_];
    __shared__ float    fsh[8];
    __shared__ int      ish[8];

    const int q = blockIdx.x;
    const int h = blockIdx.y;
    const int b = blockIdx.z;
    const int tid = threadIdx.x;
    float*       row  = scores + (((size_t)(b * H_ + h)) * S_ + q) * S_;
    const float* mrow = maskfb + (((size_t)(b * H_ + h)) * S_ + q) * S_;

    for (int k = tid; k < S_; k += 256) {
        float v = row[k];
        srow[k] = v;
        unsigned u = __float_as_uint(v);
        urow[k] = (u & 0x80000000u) ? ~u : (u | 0x80000000u);
    }
    __syncthreads();

    const int domask = (fcp[0] == 1) && (lnp[0] == 0);

    if (domask) {
        // threshold t = TOPK_-th largest key: max t with count(u >= t) >= K
        unsigned t = 0;
        for (int bit = 31; bit >= 0; --bit) {
            unsigned cand = t | (1u << bit);
            int local = 0;
            for (int k = tid; k < S_; k += 256)
                local += (urow[k] >= cand) ? 1 : 0;
            int cnt = block_sum_i(local, ish, tid);
            if (cnt >= TOPK_) t = cand;
        }
        // scores *= mask_fb * (score >= tk), with row 0 / col 0 forced to 1
        for (int k = tid; k < S_; k += 256) {
            float m;
            if (q == 0 || k == 0) m = 1.0f;
            else m = mrow[k] * ((urow[k] >= t) ? 1.0f : 0.0f);
            srow[k] = srow[k] * m;
        }
        __syncthreads();
    }

    // ---- softmax over the (masked) row
    float lmax = -3.4e38f;
    for (int k = tid; k < S_; k += 256) lmax = fmaxf(lmax, srow[k]);
    float rmax = block_max_f(lmax, fsh, tid);

    float lsum = 0.f;
    for (int k = tid; k < S_; k += 256) {
        float e = expf(srow[k] - rmax);
        srow[k] = e;
        lsum += e;
    }
    float inv = 1.0f / block_sum_f(lsum, fsh, tid);

    for (int k = tid; k < S_; k += 256) row[k] = srow[k] * inv;
}

// ---------------------------------------------------------------------------
extern "C" void kernel_launch(void* const* d_in, const int* in_sizes, int n_in,
                              void* d_out, int out_size, void* d_ws, size_t ws_size,
                              hipStream_t stream)
{
    const float* X   = (const float*)d_in[0];   // hidden_states (B,S,D)
    const float* mfb = (const float*)d_in[1];   // mask_fb (B,H,S,S)
    const float* Wq  = (const float*)d_in[2];
    const float* bq  = (const float*)d_in[3];
    const float* Wk  = (const float*)d_in[4];
    const float* bk  = (const float*)d_in[5];
    const float* Wv  = (const float*)d_in[6];
    const float* bv  = (const float*)d_in[7];
    const float* Wo  = (const float*)d_in[8];
    const float* bo  = (const float*)d_in[9];
    const int*   fc  = (const int*)d_in[10];    // forward_counts
    const int*   ln  = (const int*)d_in[11];    // layer_num
    (void)in_sizes; (void)n_in; (void)out_size; (void)ws_size;

    float* outAttn = (float*)d_out;                           // (B,S,D)
    float* outW    = outAttn + (size_t)B_ * S_ * D_;          // (B,H,S,S)

    const size_t MD = (size_t)B_ * S_ * D_;
    float* Q   = (float*)d_ws;       // (B,S,D), head-major via strided view
    float* Km  = Q   + MD;
    float* V   = Km  + MD;
    float* CTX = V   + MD;

    const int M = B_ * S_;           // 6280 rows
    dim3 blk(256, 1, 1);
    dim3 gProj((D_ + 63) / 64, (M + 127) / 128, 1);

    // Q/K/V projections: Y = X @ W + b
    wmma_gemm<<<gProj, blk, 0, stream>>>(X, Wq, bq, Q,  M, D_, D_, D_, D_, D_,
                                         0, 0, 0, 0, 0, 0, 1, 1.0f, 0);
    wmma_gemm<<<gProj, blk, 0, stream>>>(X, Wk, bk, Km, M, D_, D_, D_, D_, D_,
                                         0, 0, 0, 0, 0, 0, 1, 1.0f, 0);
    wmma_gemm<<<gProj, blk, 0, stream>>>(X, Wv, bv, V,  M, D_, D_, D_, D_, D_,
                                         0, 0, 0, 0, 0, 0, 1, 1.0f, 0);

    // scores[b,h] = (q_bh @ k_bh^T) / 8   -> written straight into weights out
    dim3 gS((S_ + 63) / 64, (S_ + 127) / 128, B_ * H_);
    wmma_gemm<<<gS, blk, 0, stream>>>(Q, Km, nullptr, outW,
                                      S_, S_, HD_, D_, D_, S_,
                                      (long long)S_ * D_, HD_,
                                      (long long)S_ * D_, HD_,
                                      (long long)H_ * S_ * S_, (long long)S_ * S_,
                                      H_, 0.125f, 1);

    // top-k threshold + mask + softmax, in place (probs == weights output)
    dim3 gT(S_, H_, B_);
    topk_softmax<<<gT, blk, 0, stream>>>(outW, mfb, fc, ln);

    // ctx[b,h] = probs_bh @ v_bh
    dim3 gC(1, (S_ + 127) / 128, B_ * H_);
    wmma_gemm<<<gC, blk, 0, stream>>>(outW, V, nullptr, CTX,
                                      (long long)S_, HD_, S_, S_, D_, D_,
                                      (long long)H_ * S_ * S_, (long long)S_ * S_,
                                      (long long)S_ * D_, HD_,
                                      (long long)S_ * D_, HD_,
                                      H_, 1.0f, 0);

    // attention_output = ctx @ Wo + bo
    wmma_gemm<<<gProj, blk, 0, stream>>>(CTX, Wo, bo, outAttn,
                                         M, D_, D_, D_, D_, D_,
                                         0, 0, 0, 0, 0, 0, 1, 1.0f, 0);
}
